// SelfAttention_22119081574652
// MI455X (gfx1250) — matrix operand order repaired and checked by reading, not compiled, not run on hardware
//
#include <hip/hip_runtime.h>

// ---------------------------------------------------------------------------
// SelfAttention with random-sampled pre-summed K/V (B=4,S=4096,H=1024,N=16,R=2)
//
// MI455X plan:
//  * 3 QKV GEMMs (103 GFLOP) dominate -> bf16 WMMA (v_wmma_f32_16x16x32_bf16),
//    fp32 accumulation.
//  * One-time f32->bf16 pre-pass for X and (transposed) W so the GEMM hot loop
//    moves raw bf16 bytes only: tiles staged with global_load_async_to_lds_b128
//    (ASYNCcnt) into double-buffered LDS, overlapping copy with WMMA.
//  * Fragment reads index the __shared__ array DIRECTLY (constant per-parity
//    offsets) so clang keeps addrspace(3) inference -> ds_load_b128. (Round 2's
//    pointer-array indirection broke this and produced global_load fragments.)
//  * 128x128 tile / 8 waves: 8 WMMAs per wave per K-step.
// ---------------------------------------------------------------------------

typedef __attribute__((ext_vector_type(16))) __bf16 v16bf;
typedef __attribute__((ext_vector_type(8)))  __bf16 bf16x8;
typedef __attribute__((ext_vector_type(8)))  float  v8f;

#define HDIM   1024
#define NHEAD  16
#define DHEAD  64
#define BATCH  4
#define SEQ    4096
#define RSAMP  2
#define ROWS   (BATCH * SEQ)       // 16384

#define TM 128
#define TN 128
#define TK 32
#define NK (HDIM / TK)             // 32 K-steps
#define XSTR 40                    // padded LDS row stride (ushorts) = 80B
#define WSTR 40
#define XBUF (TM * XSTR)
#define WBUF (TN * WSTR)
#define X0OFF 0
#define X1OFF XBUF
#define W0OFF (2 * XBUF)
#define W1OFF (2 * XBUF + WBUF)

__device__ __forceinline__ unsigned short f2bf(float f) {
  unsigned u = __builtin_bit_cast(unsigned, f);
  u += 0x7FFFu + ((u >> 16) & 1u);            // round-to-nearest-even
  return (unsigned short)(u >> 16);
}

union ABFrag { v16bf v; bf16x8 h[2]; };

// low 32 bits of a generic pointer to __shared__ == wave-relative LDS offset
__device__ __forceinline__ unsigned lds_off32(const void* p) {
  return (unsigned)(unsigned long long)p;
}

// 16B global -> LDS async copy (GLOBAL_LOAD_ASYNC_TO_LDS_B128, ASYNCcnt)
__device__ __forceinline__ void async_cp16(unsigned lds_byte_off,
                                           const unsigned short* g) {
  asm volatile("global_load_async_to_lds_b128 %0, %1, off"
               :: "v"(lds_byte_off), "v"(g) : "memory");
}
__device__ __forceinline__ void wait_async0() {
  asm volatile("s_wait_asynccnt 0x0" ::: "memory");
}

// ---------------------------------------------------------------------------
// Pre-pass A: streaming f32 -> bf16 (8 elements / thread)
// ---------------------------------------------------------------------------
__global__ __launch_bounds__(256)
void convert_kernel(const float* __restrict__ in, unsigned short* __restrict__ out) {
  const size_t t = (size_t)blockIdx.x * 256 + threadIdx.x;
  const float4 a = *reinterpret_cast<const float4*>(in + t * 8);
  const float4 b = *reinterpret_cast<const float4*>(in + t * 8 + 4);
  unsigned short o[8] = { f2bf(a.x), f2bf(a.y), f2bf(a.z), f2bf(a.w),
                          f2bf(b.x), f2bf(b.y), f2bf(b.z), f2bf(b.w) };
  *reinterpret_cast<uint4*>(out + t * 8) = *reinterpret_cast<const uint4*>(o);
}

// ---------------------------------------------------------------------------
// Pre-pass B: W[k][n] (f32) -> Wt[n][k] (bf16) via 32x32 LDS-tiled transpose
// ---------------------------------------------------------------------------
__global__ __launch_bounds__(256)
void convert_transpose_kernel(const float* __restrict__ W,
                              unsigned short* __restrict__ Wt) {
  __shared__ unsigned short tile[32][34];
  const int k0 = blockIdx.x * 32;
  const int n0 = blockIdx.y * 32;
  const int tid = threadIdx.x;
  const int r = tid >> 3, c = (tid & 7) * 4;
  const float4 wv = *reinterpret_cast<const float4*>(
      W + (size_t)(k0 + r) * HDIM + n0 + c);
  tile[r][c + 0] = f2bf(wv.x);
  tile[r][c + 1] = f2bf(wv.y);
  tile[r][c + 2] = f2bf(wv.z);
  tile[r][c + 3] = f2bf(wv.w);
  __syncthreads();
  unsigned short o[4] = { tile[c + 0][r], tile[c + 1][r],
                          tile[c + 2][r], tile[c + 3][r] };
  *reinterpret_cast<uint2*>(Wt + (size_t)(n0 + r) * HDIM + k0 + c) =
      *reinterpret_cast<const uint2*>(o);
}

// ---------------------------------------------------------------------------
// Stage 1: out[m][n] = sum_k X[m][k]*W[k][n] + bias[n]
// ---------------------------------------------------------------------------
__global__ __launch_bounds__(256)
void qkv_gemm_kernel(const unsigned short* __restrict__ Xbf,
                     const unsigned short* __restrict__ Wt,
                     const float* __restrict__ bias,
                     float* __restrict__ out) {
  __shared__ alignas(16) unsigned short smem[2 * XBUF + 2 * WBUF];

  const int tid  = threadIdx.x;
  const int lane = tid & 31;
  const int wave = tid >> 5;
  const int wm   = wave & 3;
  const int wn   = wave >> 2;
  const int mBase = blockIdx.x * TM;
  const int nBase = blockIdx.y * TN;

  const unsigned ldsBase = lds_off32(smem);

  // Async-copy chunk assignment: 512 16B chunks per tile, 2 per thread.
  const int xr0 = tid >> 2,        xcc = (tid & 3) * 8;
  const int xr1 = xr0 + 64;
  const unsigned xo0 = (unsigned)(xr0 * XSTR + xcc) * 2;
  const unsigned xo1 = (unsigned)(xr1 * XSTR + xcc) * 2;
  const unsigned short* xp0 = Xbf + (size_t)(mBase + xr0) * HDIM + xcc;
  const unsigned short* xp1 = Xbf + (size_t)(mBase + xr1) * HDIM + xcc;
  const unsigned short* wp0 = Wt + (size_t)(nBase + xr0) * HDIM + xcc;
  const unsigned short* wp1 = Wt + (size_t)(nBase + xr1) * HDIM + xcc;

  // Fragment coordinates (CDNA5 v_wmma_f32_16x16x32_bf16 layouts).
  const int arow = lane & 15;
  const int akb  = (lane >> 4) << 3;          // 0 | 8
  const int bcol = lane & 15;
  const int bkb  = (lane >> 4) << 3;          // 0 | 8
  int aBase[2], bBase[4];
  #pragma unroll
  for (int mt = 0; mt < 2; ++mt)
    aBase[mt] = (wm * 32 + mt * 16 + arow) * XSTR + akb;
  #pragma unroll
  for (int nt = 0; nt < 4; ++nt)
    bBase[nt] = (wn * 64 + nt * 16 + bcol) * WSTR + bkb;

  v8f acc[2][4] = {};

  // Prologue: stage K-slab 0 into buffer parity 0 (X/W share chunk geometry).
  async_cp16(ldsBase + (unsigned)(X0OFF * 2) + xo0, xp0);
  async_cp16(ldsBase + (unsigned)(X0OFF * 2) + xo1, xp1);
  async_cp16(ldsBase + (unsigned)(W0OFF * 2) + xo0, wp0);
  async_cp16(ldsBase + (unsigned)(W0OFF * 2) + xo1, wp1);

  #pragma unroll 2
  for (int it = 0; it < NK; ++it) {
    const int cur = it & 1;
    const int xCur = cur ? X1OFF : X0OFF;
    const int wCur = cur ? W1OFF : W0OFF;
    const int xNxt = cur ? X0OFF : X1OFF;
    const int wNxt = cur ? W0OFF : W1OFF;

    wait_async0();                 // my copies into buf[cur] have landed
    __syncthreads();               // everyone's landed; prior reads drained

    if (it + 1 < NK) {
      xp0 += TK; xp1 += TK; wp0 += TK; wp1 += TK;
      async_cp16(ldsBase + (unsigned)(xNxt * 2) + xo0, xp0);
      async_cp16(ldsBase + (unsigned)(xNxt * 2) + xo1, xp1);
      async_cp16(ldsBase + (unsigned)(wNxt * 2) + xo0, wp0);
      async_cp16(ldsBase + (unsigned)(wNxt * 2) + xo1, wp1);
    }

    ABFrag a[2], b[4];
    #pragma unroll
    for (int mt = 0; mt < 2; ++mt) {
      a[mt].h[0] = *reinterpret_cast<const bf16x8*>(&smem[xCur + aBase[mt]]);
      a[mt].h[1] = *reinterpret_cast<const bf16x8*>(&smem[xCur + aBase[mt] + 16]);
    }
    #pragma unroll
    for (int nt = 0; nt < 4; ++nt) {
      b[nt].h[0] = *reinterpret_cast<const bf16x8*>(&smem[wCur + bBase[nt]]);
      b[nt].h[1] = *reinterpret_cast<const bf16x8*>(&smem[wCur + bBase[nt] + 16]);
    }
    #pragma unroll
    for (int mt = 0; mt < 2; ++mt)
      #pragma unroll
      for (int nt = 0; nt < 4; ++nt)
        acc[mt][nt] = __builtin_amdgcn_wmma_f32_16x16x32_bf16(
            false, a[mt].v, false, b[nt].v, (short)0, acc[mt][nt], false, false);
  }

  // Epilogue: C layout VGPR g -> (row g | g+8), col = lane&15; add bias.
  const int crow = (lane >> 4) * 8;
  const int ccol = lane & 15;
  #pragma unroll
  for (int mt = 0; mt < 2; ++mt)
    #pragma unroll
    for (int nt = 0; nt < 4; ++nt) {
      const int n = nBase + wn * 64 + nt * 16 + ccol;
      const float bv = bias[n];
      float* op = out + (size_t)(mBase + wm * 32 + mt * 16 + crow) * HDIM + n;
      #pragma unroll
      for (int g = 0; g < 8; ++g)
        op[(size_t)g * HDIM] = acc[mt][nt][g] + bv;
    }
}

// ---------------------------------------------------------------------------
// Stage 2: k_sum[b,r,:] = sum_s k[b, idx[b,s,r], :]  (and v_sum); L2-resident.
// ---------------------------------------------------------------------------
__global__ __launch_bounds__(256)
void gather_sum_kernel(const float* __restrict__ kmat,
                       const float* __restrict__ vmat,
                       const int* __restrict__ idx,
                       float* __restrict__ ksum,
                       float* __restrict__ vsum) {
  const int br = blockIdx.x;
  const int b  = br / RSAMP;
  const int r  = br % RSAMP;
  const int h  = blockIdx.y * 256 + threadIdx.x;
  const int* ip = idx + (size_t)b * SEQ * RSAMP + r;
  const size_t rowbase = (size_t)b * SEQ * HDIM;
  float ak = 0.f, av = 0.f;
  for (int s = 0; s < SEQ; ++s) {
    const int j = ip[(size_t)s * RSAMP];
    const size_t off = rowbase + (size_t)j * HDIM + h;
    ak += kmat[off];
    av += vmat[off];
  }
  ksum[(size_t)br * HDIM + h] = ak;
  vsum[(size_t)br * HDIM + h] = av;
}

// ---------------------------------------------------------------------------
// Stage 3: per (b,s,n): 2 dots over D=64, softmax over R=2, combine v_sum.
// ---------------------------------------------------------------------------
__global__ __launch_bounds__(256)
void attn_kernel(const float* __restrict__ q,
                 const float* __restrict__ ksum,
                 const float* __restrict__ vsum,
                 float* __restrict__ out) {
  const int g   = blockIdx.x * 256 + threadIdx.x;
  const int n   = g & (NHEAD - 1);
  const int row = g >> 4;
  const int b   = row >> 12;
  const float4* qp  = reinterpret_cast<const float4*>(q + (size_t)row * HDIM + n * DHEAD);
  const float4* k0p = reinterpret_cast<const float4*>(ksum + (size_t)(b * RSAMP + 0) * HDIM + n * DHEAD);
  const float4* k1p = reinterpret_cast<const float4*>(ksum + (size_t)(b * RSAMP + 1) * HDIM + n * DHEAD);
  float s0 = 0.f, s1 = 0.f;
  #pragma unroll
  for (int d = 0; d < DHEAD / 4; ++d) {
    const float4 qv = qp[d], a = k0p[d], c = k1p[d];
    s0 += qv.x * a.x + qv.y * a.y + qv.z * a.z + qv.w * a.w;
    s1 += qv.x * c.x + qv.y * c.y + qv.z * c.z + qv.w * c.w;
  }
  s0 *= 0.125f;
  s1 *= 0.125f;
  const float m  = fmaxf(s0, s1);
  const float e0 = __expf(s0 - m);
  const float e1 = __expf(s1 - m);
  const float inv = 1.f / (e0 + e1);
  const float p0 = e0 * inv, p1 = e1 * inv;
  const float4* v0p = reinterpret_cast<const float4*>(vsum + (size_t)(b * RSAMP + 0) * HDIM + n * DHEAD);
  const float4* v1p = reinterpret_cast<const float4*>(vsum + (size_t)(b * RSAMP + 1) * HDIM + n * DHEAD);
  float4* op = reinterpret_cast<float4*>(out + (size_t)row * HDIM + n * DHEAD);
  #pragma unroll
  for (int d = 0; d < DHEAD / 4; ++d) {
    const float4 a = v0p[d], c = v1p[d];
    float4 o;
    o.x = p0 * a.x + p1 * c.x;
    o.y = p0 * a.y + p1 * c.y;
    o.z = p0 * a.z + p1 * c.z;
    o.w = p0 * a.w + p1 * c.w;
    op[d] = o;
  }
}

// ---------------------------------------------------------------------------
extern "C" void kernel_launch(void* const* d_in, const int* in_sizes, int n_in,
                              void* d_out, int out_size, void* d_ws, size_t ws_size,
                              hipStream_t stream) {
  const float* X  = (const float*)d_in[0];
  const float* Wq = (const float*)d_in[2];
  const float* bq = (const float*)d_in[3];
  const float* Wk = (const float*)d_in[4];
  const float* bk = (const float*)d_in[5];
  const float* Wv = (const float*)d_in[6];
  const float* bv = (const float*)d_in[7];
  const int*  idx = (const int*)d_in[8];

  float* ws   = (float*)d_ws;
  float* q    = ws;
  float* k    = ws + (size_t)ROWS * HDIM;
  float* v    = ws + 2 * (size_t)ROWS * HDIM;
  float* ksum = ws + 3 * (size_t)ROWS * HDIM;
  float* vsum = ksum + (size_t)BATCH * RSAMP * HDIM;
  unsigned short* Xbf = (unsigned short*)(vsum + (size_t)BATCH * RSAMP * HDIM);
  unsigned short* Wtq = Xbf + (size_t)ROWS * HDIM;
  unsigned short* Wtk = Wtq + (size_t)HDIM * HDIM;
  unsigned short* Wtv = Wtk + (size_t)HDIM * HDIM;

  convert_kernel<<<(size_t)ROWS * HDIM / (256 * 8), 256, 0, stream>>>(X, Xbf);
  const dim3 tGrid(HDIM / 32, HDIM / 32);
  convert_transpose_kernel<<<tGrid, 256, 0, stream>>>(Wq, Wtq);
  convert_transpose_kernel<<<tGrid, 256, 0, stream>>>(Wk, Wtk);
  convert_transpose_kernel<<<tGrid, 256, 0, stream>>>(Wv, Wtv);

  const dim3 gemmGrid(ROWS / TM, HDIM / TN);
  qkv_gemm_kernel<<<gemmGrid, 256, 0, stream>>>(Xbf, Wtq, bq, q);
  qkv_gemm_kernel<<<gemmGrid, 256, 0, stream>>>(Xbf, Wtk, bk, k);
  qkv_gemm_kernel<<<gemmGrid, 256, 0, stream>>>(Xbf, Wtv, bv, v);

  gather_sum_kernel<<<dim3(BATCH * RSAMP, HDIM / 256), 256, 0, stream>>>(
      k, v, idx, ksum, vsum);

  attn_kernel<<<(ROWS * NHEAD) / 256, 256, 0, stream>>>(
      q, ksum, vsum, (float*)d_out);
}
